// SessionGNN_24060406792471
// MI455X (gfx1250) — matrix operand order repaired and checked by reading, not compiled, not run on hardware
//
#include <hip/hip_runtime.h>
#include <hip/hip_bf16.h>
#include <math.h>

// Problem constants (from reference): B=1024, L=200, H=128, STEP=1
#define BATCH 1024
#define LSEQ  200
#define LPAD  224   // L padded up to multiple of 32 for WMMA K loops
#define HDIM  128

typedef __attribute__((ext_vector_type(16))) __bf16 v16bf;
typedef __attribute__((ext_vector_type(8)))  __bf16 v8bf;
typedef __attribute__((ext_vector_type(4)))  __bf16 v4bf;
typedef __attribute__((ext_vector_type(8)))  float  v8f;

// ---------------------------------------------------------------------------
// WMMA helpers (CDNA5 wave32, 16x16x32 bf16 -> f32)
// ---------------------------------------------------------------------------
__device__ __forceinline__ v8f wmma_bf16(v16bf a, v16bf b, v8f c) {
  // (neg_a, A, neg_b, B, c_mod, C, reuse_a, reuse_b)
  return __builtin_amdgcn_wmma_f32_16x16x32_bf16(false, a, false, b, (short)0, c,
                                                 false, false);
}

// A-matrix fragment: 16x32 bf16 slice at column offset kk of a row-major
// [16][ldk] bf16 tile. Contiguous K order (same as b_frag): lane m = lane&15; lanes<16 hold
// K = kk+{0..15}, lanes>=16 hold K = kk+{16..31}.
__device__ __forceinline__ v16bf a_frag(const __bf16* sA, int ldk, int kk) {
  const int lane = threadIdx.x & 31;
  const int m    = lane & 15;
  const int k0   = kk + ((lane & 16) ? 16 : 0);
  const __bf16* p = sA + m * ldk + k0;
  v16bf a;
#pragma unroll
  for (int i = 0; i < 8; ++i) { a[i] = p[i]; a[8 + i] = p[8 + i]; }
  return a;
}

// B-matrix fragment: B[k][n] = W[nBase+n][kk+k] with W row-major (ld = ldk).
// Contiguous K order (same as a_frag): lane holds column n = lane&15, 16 contiguous K values,
// lanes>=16 at K offset +16.  Single 32-byte per-lane load.
__device__ __forceinline__ v16bf b_frag(const __bf16* gW, int ldk, int nBase,
                                        int kk) {
  const int lane = threadIdx.x & 31;
  const int n    = nBase + (lane & 15);
  const int k0   = kk + ((lane & 16) ? 16 : 0);
  return *(const v16bf*)(gW + (size_t)n * ldk + k0);
}

__device__ __forceinline__ v4bf cvt4(float4 v) {
  v4bf r;
  r[0] = (__bf16)v.x; r[1] = (__bf16)v.y; r[2] = (__bf16)v.z; r[3] = (__bf16)v.w;
  return r;
}

// ---------------------------------------------------------------------------
// Kernel 0: f32 -> bf16 conversion (weights)
// ---------------------------------------------------------------------------
__global__ void cvt_f32_bf16(const float* __restrict__ src,
                             __bf16* __restrict__ dst, int n) {
  int i = blockIdx.x * blockDim.x + threadIdx.x;
  if (i < n) dst[i] = (__bf16)src[i];
}

// ---------------------------------------------------------------------------
// Kernel 1: e_in = h @ W_ein^T + b_ein ; e_out = h @ W_eout^T + b_eout
// Stored TRANSPOSED as eT[b][hcol (128)][l (LPAD)] in bf16, rows l>=LSEQ
// zero-filled so padded-K WMMA contributions vanish in the next pass.
// Grid: (BATCH, LPAD/16). Block: 256 (8 waves); wave w -> N-tile w.
// ---------------------------------------------------------------------------
__global__ __launch_bounds__(256) void e_kernel(
    const float* __restrict__ h, const __bf16* __restrict__ Wein16,
    const __bf16* __restrict__ Weout16, const float* __restrict__ b_ein,
    const float* __restrict__ b_eout, __bf16* __restrict__ eInT,
    __bf16* __restrict__ eOutT) {
  __shared__ __align__(16) __bf16 sH[16 * HDIM];

  const int b     = blockIdx.x;
  const int lBase = blockIdx.y * 16;
  const int tid   = threadIdx.x;

  // Stage h tile (f32 -> bf16), zero rows beyond L.  16 rows x 32 float4.
  for (int idx = tid; idx < 16 * 32; idx += 256) {
    const int r = idx >> 5, q = idx & 31;
    const int l = lBase + r;
    float4 v = {0.0f, 0.0f, 0.0f, 0.0f};
    if (l < LSEQ)
      v = ((const float4*)(h + ((size_t)b * LSEQ + l) * HDIM))[q];
    *(v4bf*)(sH + r * HDIM + q * 4) = cvt4(v);
  }
  __syncthreads();

  const int wave  = tid >> 5;
  const int lane  = tid & 31;
  const int nBase = wave * 16;

  v8f accIn = {}, accOut = {};
#pragma unroll
  for (int kk = 0; kk < HDIM; kk += 32) {
    const v16bf a  = a_frag(sH, HDIM, kk);
    const v16bf bi = b_frag(Wein16, HDIM, nBase, kk);
    const v16bf bo = b_frag(Weout16, HDIM, nBase, kk);
    accIn  = wmma_bf16(a, bi, accIn);
    accOut = wmma_bf16(a, bo, accOut);
  }

  const int   n      = nBase + (lane & 15);
  const float biasI  = b_ein[n];
  const float biasO  = b_eout[n];
  const int   rowOff = (lane & 16) ? 8 : 0;

  v8bf pi, po;
#pragma unroll
  for (int i = 0; i < 8; ++i) {
    const int l = lBase + rowOff + i;
    pi[i] = (l < LSEQ) ? (__bf16)(accIn[i] + biasI) : (__bf16)0.0f;
    po[i] = (l < LSEQ) ? (__bf16)(accOut[i] + biasO) : (__bf16)0.0f;
  }
  const size_t off = ((size_t)b * HDIM + n) * LPAD + lBase + rowOff;
  *(v8bf*)(eInT + off)  = pi;   // 16B contiguous store (transposed layout)
  *(v8bf*)(eOutT + off) = po;
}

// ---------------------------------------------------------------------------
// Kernel 2 (fused): m_in/m_out batched GEMMs + gate GEMMs + GRU gating.
// Grid: (BATCH, ceil(L/16)=13). Block: 256 (8 waves).
// Phase 1: wave w computes 16x16 tiles (cols 16w..) of m_in and m_out,
//          writes bias-added bf16 [m_in|m_out] (16x256) to LDS.
// Phase 2: wave w computes gate columns 16w.. : accR/accI accumulate both
//          w_ih and w_hh GEMMs; i_n / h_n kept separate; then gating.
// ---------------------------------------------------------------------------
__global__ __launch_bounds__(256) void fused_kernel(
    const float* __restrict__ A, const float* __restrict__ h,
    const __bf16* __restrict__ wih16, const __bf16* __restrict__ whh16,
    const float* __restrict__ b_ih, const float* __restrict__ b_hh,
    const float* __restrict__ b_iah, const float* __restrict__ b_oah,
    const __bf16* __restrict__ eInT, const __bf16* __restrict__ eOutT,
    float* __restrict__ out) {
  __shared__ __align__(16) __bf16 sAin[16 * LPAD];
  __shared__ __align__(16) __bf16 sAout[16 * LPAD];
  __shared__ __align__(16) __bf16 sIn[16 * 256];   // concat(m_in, m_out) bf16
  __shared__ __align__(16) __bf16 sHb[16 * HDIM];  // h tile bf16 (gh GEMM)
  __shared__ __align__(16) float  sHf[16 * HDIM];  // h tile f32  (gating)

  const int b     = blockIdx.x;
  const int lBase = blockIdx.y * 16;
  const int tid   = threadIdx.x;
  const float* Ab = A + (size_t)b * LSEQ * (2 * LSEQ);

  const __bf16* eIb = eInT + (size_t)b * HDIM * LPAD;
  const __bf16* eOb = eOutT + (size_t)b * HDIM * LPAD;

  // Prefetch this batch's e slices (phase-1 B operands) while staging runs.
  // 128*224 bf16 = 57344 B = 896 x 64B lines; 256 threads x 4 lines.
#pragma unroll
  for (int j = 0; j < 4; ++j) {
    const int e = (tid * 4 + j) * 32;  // 32 bf16 = one 64B line
    if (e < HDIM * LPAD) {
      __builtin_prefetch(eIb + e, 0, 3);
      __builtin_prefetch(eOb + e, 0, 3);
    }
  }

  // Stage a_in / a_out tiles (f32 -> bf16).  Row = 56 float4 slots (224
  // cols); slots 0..49 are in-bounds (LSEQ=200 divisible by 4), 50..55 pad.
  for (int idx = tid; idx < 16 * 56; idx += 256) {
    const int r = idx / 56, q = idx % 56;
    const int l = lBase + r;
    float4 vi = {0.0f, 0.0f, 0.0f, 0.0f};
    float4 vo = {0.0f, 0.0f, 0.0f, 0.0f};
    if (l < LSEQ && q < 50) {
      const float4* p = (const float4*)(Ab + (size_t)l * (2 * LSEQ));
      vi = p[q];        // a_in [k .. k+3]
      vo = p[50 + q];   // a_out[k .. k+3] (offset 200 floats = 50 float4)
    }
    *(v4bf*)(sAin + r * LPAD + q * 4)  = cvt4(vi);
    *(v4bf*)(sAout + r * LPAD + q * 4) = cvt4(vo);
  }
  // Stage h tile (both precisions), 16 rows x 32 float4
  for (int idx = tid; idx < 16 * 32; idx += 256) {
    const int r = idx >> 5, q = idx & 31;
    const int l = lBase + r;
    float4 v = {0.0f, 0.0f, 0.0f, 0.0f};
    if (l < LSEQ)
      v = ((const float4*)(h + ((size_t)b * LSEQ + l) * HDIM))[q];
    ((float4*)sHf)[idx] = v;
    *(v4bf*)(sHb + r * HDIM + q * 4) = cvt4(v);
  }
  __syncthreads();

  const int wave = tid >> 5;
  const int lane = tid & 31;

  // ---- Phase 1: m_in / m_out ----
  {
    const int nBase = wave * 16;
    v8f mi = {}, mo = {};
#pragma unroll
    for (int kk = 0; kk < LPAD; kk += 32) {
      const v16bf ai = a_frag(sAin, LPAD, kk);
      const v16bf ao = a_frag(sAout, LPAD, kk);
      const v16bf bi = b_frag(eIb, LPAD, nBase, kk);
      const v16bf bo = b_frag(eOb, LPAD, nBase, kk);
      mi = wmma_bf16(ai, bi, mi);
      mo = wmma_bf16(ao, bo, mo);
    }
    const int   n      = nBase + (lane & 15);
    const float biasI  = b_iah[n];
    const float biasO  = b_oah[n];
    const int   rowOff = (lane & 16) ? 8 : 0;
#pragma unroll
    for (int i = 0; i < 8; ++i) {
      const int m = rowOff + i;
      sIn[m * 256 + n]        = (__bf16)(mi[i] + biasI);
      sIn[m * 256 + HDIM + n] = (__bf16)(mo[i] + biasO);
    }
  }
  __syncthreads();

  // ---- Phase 2: gates ----
  const int nR = wave * 16;  // this wave's H-column tile
  v8f accR = {}, accI = {}, accIN = {}, accHN = {};
#pragma unroll
  for (int kk = 0; kk < 256; kk += 32) {  // gi = [m_in|m_out] @ w_ih^T
    const v16bf a = a_frag(sIn, 256, kk);
    accR  = wmma_bf16(a, b_frag(wih16, 256, nR, kk), accR);
    accI  = wmma_bf16(a, b_frag(wih16, 256, HDIM + nR, kk), accI);
    accIN = wmma_bf16(a, b_frag(wih16, 256, 2 * HDIM + nR, kk), accIN);
  }
#pragma unroll
  for (int kk = 0; kk < HDIM; kk += 32) {  // gh = h @ w_hh^T
    const v16bf a = a_frag(sHb, HDIM, kk);
    accR  = wmma_bf16(a, b_frag(whh16, HDIM, nR, kk), accR);        // i_r+h_r
    accI  = wmma_bf16(a, b_frag(whh16, HDIM, HDIM + nR, kk), accI); // i_i+h_i
    accHN = wmma_bf16(a, b_frag(whh16, HDIM, 2 * HDIM + nR, kk), accHN);
  }

  const int   n   = nR + (lane & 15);
  const float br  = b_ih[n] + b_hh[n];
  const float bz  = b_ih[HDIM + n] + b_hh[HDIM + n];
  const float bin = b_ih[2 * HDIM + n];
  const float bhn = b_hh[2 * HDIM + n];
  const int rowOff = (lane & 16) ? 8 : 0;
#pragma unroll
  for (int i = 0; i < 8; ++i) {
    const int m = rowOff + i;
    const int l = lBase + m;
    const float r  = 1.0f / (1.0f + __expf(-(accR[i] + br)));
    const float z  = 1.0f / (1.0f + __expf(-(accI[i] + bz)));
    const float ng = tanhf(accIN[i] + bin + r * (accHN[i] + bhn));
    const float hv = sHf[m * HDIM + n];
    if (l < LSEQ) out[((size_t)b * LSEQ + l) * HDIM + n] = ng + z * (hv - ng);
  }
}

// ---------------------------------------------------------------------------
// Launch
// ---------------------------------------------------------------------------
extern "C" void kernel_launch(void* const* d_in, const int* in_sizes, int n_in,
                              void* d_out, int out_size, void* d_ws,
                              size_t ws_size, hipStream_t stream) {
  (void)in_sizes; (void)n_in; (void)out_size; (void)ws_size;

  const float* A      = (const float*)d_in[0];
  const float* h      = (const float*)d_in[1];
  const float* w_ih   = (const float*)d_in[2];
  const float* w_hh   = (const float*)d_in[3];
  const float* b_ih   = (const float*)d_in[4];
  const float* b_hh   = (const float*)d_in[5];
  const float* b_iah  = (const float*)d_in[6];
  const float* b_oah  = (const float*)d_in[7];
  const float* W_ein  = (const float*)d_in[8];
  const float* b_ein  = (const float*)d_in[9];
  const float* W_eout = (const float*)d_in[10];
  const float* b_eout = (const float*)d_in[11];
  float* out = (float*)d_out;

  // Workspace layout (bytes)
  char* ws = (char*)d_ws;
  const size_t eElems = (size_t)BATCH * HDIM * LPAD;  // 29,360,128
  __bf16* eInT    = (__bf16*)(ws);
  __bf16* eOutT   = (__bf16*)(ws + eElems * 2);
  __bf16* wih16   = (__bf16*)(ws + eElems * 4);
  __bf16* whh16   = (__bf16*)(ws + eElems * 4 + 196608);
  __bf16* wein16  = (__bf16*)(ws + eElems * 4 + 196608 + 98304);
  __bf16* weout16 = (__bf16*)(ws + eElems * 4 + 196608 + 98304 + 32768);

  // Weight conversions (tiny; L2-resident afterwards)
  cvt_f32_bf16<<<dim3((98304 + 255) / 256), 256, 0, stream>>>(w_ih, wih16, 98304);
  cvt_f32_bf16<<<dim3((49152 + 255) / 256), 256, 0, stream>>>(w_hh, whh16, 49152);
  cvt_f32_bf16<<<dim3((16384 + 255) / 256), 256, 0, stream>>>(W_ein, wein16, 16384);
  cvt_f32_bf16<<<dim3((16384 + 255) / 256), 256, 0, stream>>>(W_eout, weout16, 16384);

  // e_in/e_out (stored transposed+padded bf16; both fit in 192MB L2)
  e_kernel<<<dim3(BATCH, LPAD / 16), 256, 0, stream>>>(h, wein16, weout16,
                                                       b_ein, b_eout, eInT,
                                                       eOutT);

  // Fused propagation + GRU gating
  fused_kernel<<<dim3(BATCH, (LSEQ + 15) / 16), 256, 0, stream>>>(
      A, h, wih16, whh16, b_ih, b_hh, b_iah, b_oah, eInT, eOutT, out);
}
